// RNN_85212151153260
// MI455X (gfx1250) — matrix operand order repaired and checked by reading, not compiled, not run on hardware
//
#include <hip/hip_runtime.h>

// ---------------------------------------------------------------------------
// RNN scan for MI455X (gfx1250, wave32, WMMA, TDM).
//   N=64 batch, L=2048 steps, H=256 hidden, V=32000 vocab.
//   Phase 0: convert W_xh / W_hh to f16 in workspace.
//   Phase 1: xh = emb[x] @ W_xh^T + b_xh  -> written into d_out (in-place buf)
//            embedding rows fetched via TDM gather-mode (token ids as indices)
//   Phase 2: persistent scan, 4 blocks x 16 waves, one 16x16 WMMA tile/wave,
//            W_hh fragments hoisted to VGPRs, xh[t+1] TDM-prefetched into
//            double-buffered LDS, h exchanged through LDS each step.
// ---------------------------------------------------------------------------

#define DEV __device__ __forceinline__

typedef __attribute__((ext_vector_type(16))) _Float16 v16h;
typedef __attribute__((ext_vector_type(8)))  float    v8f;
typedef unsigned int uint32x4 __attribute__((ext_vector_type(4)));
typedef int          int32x8  __attribute__((ext_vector_type(8)));
typedef int          int32x4  __attribute__((ext_vector_type(4)));

constexpr int NB   = 64;
constexpr int LSEQ = 2048;
constexpr int VOC  = 32000;
constexpr int H    = 256;
constexpr int AS   = 264;                 // ushort (f16) LDS row stride: 528B -> conflict-free b128 reads
constexpr int ES   = 260;                 // f32 LDS row stride: 1040B -> conflict-free b32 reads
constexpr unsigned long long RS = (unsigned long long)LSEQ * H;  // out row stride per batch elem

union AB { v16h v; uint4 q[2]; };

DEV unsigned short f2h(float f) {
  _Float16 h = (_Float16)f;               // v_cvt_f16_f32, RNE
  return __builtin_bit_cast(unsigned short, h);
}

DEV unsigned lds_off(const void* p) {     // low 32 bits of flat shared addr == LDS byte address
  return (unsigned)(unsigned long long)p;
}

// --- TDM: plain 2D f32 tile load, LDS rows padded +4 dwords every 256 dwords ---
DEV void tdm_tile_f32(unsigned lds, const float* g, unsigned dim0, unsigned dim1,
                      unsigned tile0, unsigned tile1, unsigned stride0) {
  unsigned long long ga = (unsigned long long)g;
  uint32x4 g0; int32x8 g1; int32x4 gz; int32x8 gz8;
  g0[0] = 1u;                                           // count=1, no gather
  g0[1] = lds;                                          // lds_addr
  g0[2] = (unsigned)ga;                                 // global_addr[31:0]
  g0[3] = (unsigned)(ga >> 32) | (2u << 30);            // global_addr[56:32] | type=2
  g1[0] = (int)((2u << 16) | (1u << 20) | (7u << 22) | (3u << 25)); // 4B elems, pad 4dw/256dw
  g1[1] = (int)(dim0 << 16);                            // tensor_dim0[15:0]
  g1[2] = (int)((dim0 >> 16) | (dim1 << 16));           // tensor_dim0[31:16] | tensor_dim1[15:0]
  g1[3] = (int)((dim1 >> 16) | (tile0 << 16));          // tensor_dim1[31:16] | tile_dim0
  g1[4] = (int)tile1;                                   // tile_dim1 (tile_dim2=0)
  g1[5] = (int)stride0;                                 // tensor_dim0_stride[31:0]
  g1[6] = 0; g1[7] = 0;
  gz[0] = 0; gz[1] = 0; gz[2] = 0; gz[3] = 0;
#pragma unroll
  for (int i = 0; i < 8; ++i) gz8[i] = 0;
#pragma unroll
  for (int i = 0; i < 4; ++i) g0[i] = (unsigned)__builtin_amdgcn_readfirstlane((int)g0[i]);
#pragma unroll
  for (int i = 0; i < 8; ++i) g1[i] = __builtin_amdgcn_readfirstlane(g1[i]);
  __builtin_amdgcn_tensor_load_to_lds(g0, g1, gz, gz, gz8, 0);
}

// --- TDM: gather-mode load of 16 f32 rows (16-bit row indices packed in pk[8]) ---
DEV void tdm_gather16_f32(unsigned lds, const float* g, unsigned dim0, unsigned dim1,
                          unsigned tile0, unsigned nidx, unsigned stride0, const int* pk) {
  unsigned long long ga = (unsigned long long)g;
  uint32x4 g0; int32x8 g1; int32x4 g2, g3; int32x8 gz8;
  g0[0] = 1u | (1u << 31);                              // count=1 | gather_mode, 16-bit idx
  g0[1] = lds;
  g0[2] = (unsigned)ga;
  g0[3] = (unsigned)(ga >> 32) | (2u << 30);
  g1[0] = (int)((2u << 16) | (1u << 20) | (7u << 22) | (3u << 25));
  g1[1] = (int)(dim0 << 16);
  g1[2] = (int)((dim0 >> 16) | (dim1 << 16));
  g1[3] = (int)((dim1 >> 16) | (tile0 << 16));
  g1[4] = (int)nidx;                                    // tile_dim1 = #valid indices
  g1[5] = (int)stride0;
  g1[6] = 0; g1[7] = 0;
#pragma unroll
  for (int i = 0; i < 8; ++i) gz8[i] = 0;
#pragma unroll
  for (int i = 0; i < 4; ++i) g0[i] = (unsigned)__builtin_amdgcn_readfirstlane((int)g0[i]);
#pragma unroll
  for (int i = 0; i < 8; ++i) g1[i] = __builtin_amdgcn_readfirstlane(g1[i]);
#pragma unroll
  for (int i = 0; i < 4; ++i) {
    g2[i] = __builtin_amdgcn_readfirstlane(pk[i]);
    g3[i] = __builtin_amdgcn_readfirstlane(pk[i + 4]);
  }
  __builtin_amdgcn_tensor_load_to_lds(g0, g1, g2, g3, gz8, 0);
}

// ---------------------------------------------------------------------------
// Phase 0: f32 -> f16 weight conversion into workspace.
// ---------------------------------------------------------------------------
__global__ __launch_bounds__(256) void cvt_f16(const float* __restrict__ s,
                                               unsigned short* __restrict__ d, int n) {
  int i = blockIdx.x * 256 + threadIdx.x;
  if (i < n) d[i] = f2h(s[i]);
}

// ---------------------------------------------------------------------------
// Phase 1: xh[m, k] = sum_h emb[x[m], h] * W_xh[k, h] + b_xh[k]   (m = n*L+l)
// One block per 16-row M tile; TDM-gathers 16 embedding rows; 8 waves x 2 tiles.
// ---------------------------------------------------------------------------
__global__ __launch_bounds__(256) void xh_gemm(const int* __restrict__ xids,
                                               const float* __restrict__ emb,
                                               const unsigned short* __restrict__ wxh,
                                               const float* __restrict__ bxh,
                                               float* __restrict__ out) {
  __shared__ float          e_lds[16 * ES];   // gathered f32 emb rows (TDM pads stride to 260)
  __shared__ unsigned short a_lds[16 * AS];   // f16 A tile

  const int tid = threadIdx.x;
  const int m0  = blockIdx.x * 16;

  if (tid < 32) {                             // wave 0 issues the TDM gather
    int pk[8];
#pragma unroll
    for (int j = 0; j < 8; ++j) {
      unsigned a = (unsigned)xids[m0 + 2 * j]     & 0xFFFFu;
      unsigned b = (unsigned)xids[m0 + 2 * j + 1] & 0xFFFFu;
      pk[j] = (int)(a | (b << 16));
    }
    tdm_gather16_f32(lds_off(e_lds), emb, H, VOC, H, 16, H, pk);
    __builtin_amdgcn_s_wait_tensorcnt(0);
  }
  __syncthreads();

  // f32 -> f16 conversion of the A tile
#pragma unroll
  for (int j = 0; j < 16; ++j)
    a_lds[j * AS + tid] = f2h(e_lds[j * ES + tid]);
  __syncthreads();

  const int wv   = tid >> 5;
  const int lane = tid & 31;
  const int hi   = lane >> 4;
  const int row  = lane & 15;

#pragma unroll
  for (int tt = 0; tt < 2; ++tt) {
    const int n0   = (wv * 2 + tt) * 16;
    const int colG = n0 + (lane & 15);
    v8f acc = {0.f, 0.f, 0.f, 0.f, 0.f, 0.f, 0.f, 0.f};
#pragma unroll
    for (int kk = 0; kk < 8; ++kk) {
      AB a, b;
      const int kA = kk * 32 + hi * 16;
      a.q[0] = *(const uint4*)(a_lds + row * AS + kA);
      a.q[1] = *(const uint4*)(a_lds + row * AS + kA + 8);
      const int kB = kk * 32 + hi * 16;
      b.q[0] = *(const uint4*)(wxh + colG * H + kB);
      b.q[1] = *(const uint4*)(wxh + colG * H + kB + 8);
      acc = __builtin_amdgcn_wmma_f32_16x16x32_f16(false, a.v, false, b.v,
                                                   (short)0, acc, false, false);
    }
    const float bias = bxh[colG];
#pragma unroll
    for (int r = 0; r < 8; ++r)
      out[(unsigned long long)(m0 + hi * 8 + r) * H + colG] = acc[r] + bias;
  }
}

// ---------------------------------------------------------------------------
// Phase 2: persistent scan. grid=4 (16 batch rows each), 512 threads = 16 waves.
// h_t = tanh(h_{t-1} @ W_hh^T + b_hh + xh_t); in-place over d_out.
// ---------------------------------------------------------------------------
__global__ __launch_bounds__(512) void rnn_scan(const unsigned short* __restrict__ whh,
                                                const float* __restrict__ bhh,
                                                float* __restrict__ out) {
  __shared__ unsigned short h_lds[16 * AS];       // current h, f16
  __shared__ float          xh_lds[2][16 * ES];   // double-buffered xh slice (TDM dest)

  const int tid  = threadIdx.x;
  const int wv   = tid >> 5;
  const int lane = tid & 31;
  const int hi   = lane >> 4;
  const int row  = lane & 15;
  const int n0   = wv * 16;
  const int colG = n0 + (lane & 15);
  const int nbase = blockIdx.x * 16;

  for (int i = tid; i < 16 * AS; i += 512) h_lds[i] = 0;   // h0 = 0

  // Hoist all W_hh B fragments for this wave's tile into VGPRs (loop-invariant).
  AB bf[8];
#pragma unroll
  for (int kk = 0; kk < 8; ++kk) {
    const int kB = kk * 32 + hi * 16;
    bf[kk].q[0] = *(const uint4*)(whh + colG * H + kB);
    bf[kk].q[1] = *(const uint4*)(whh + colG * H + kB + 8);
  }
  const float bias = bhh[colG];
  float* orow = out + (unsigned long long)(nbase + hi * 8) * RS + colG;

  if (wv == 0)   // prime the pipeline: xh[0]
    tdm_tile_f32(lds_off(xh_lds[0]), out + (unsigned long long)nbase * RS,
                 H, 16, H, 16, (unsigned)RS);

  for (int t = 0; t < LSEQ; ++t) {
    if (wv == 0) __builtin_amdgcn_s_wait_tensorcnt(0);     // xh[t] landed
    __syncthreads();                                       // everyone past step t-1
    if (wv == 0 && (t + 1) < LSEQ)                         // prefetch xh[t+1] into other buffer
      tdm_tile_f32(lds_off(xh_lds[(t + 1) & 1]),
                   out + (unsigned long long)nbase * RS + (unsigned)(t + 1) * H,
                   H, 16, H, 16, (unsigned)RS);

    const float* xh = xh_lds[t & 1];
    v8f acc;
#pragma unroll
    for (int r = 0; r < 8; ++r) acc[r] = xh[(hi * 8 + r) * ES + colG] + bias;

#pragma unroll
    for (int kk = 0; kk < 8; ++kk) {
      AB a;
      const int kA = kk * 32 + hi * 16;
      a.q[0] = *(const uint4*)(h_lds + row * AS + kA);
      a.q[1] = *(const uint4*)(h_lds + row * AS + kA + 8);
      acc = __builtin_amdgcn_wmma_f32_16x16x32_f16(false, a.v, false, bf[kk].v,
                                                   (short)0, acc, false, false);
    }

    float hv[8];
#pragma unroll
    for (int r = 0; r < 8; ++r) hv[r] = tanhf(acc[r]);
#pragma unroll
    for (int r = 0; r < 8; ++r)
      orow[(unsigned long long)r * RS + (unsigned long long)t * H] = hv[r];

    __syncthreads();                                       // all A reads of h_lds done
#pragma unroll
    for (int r = 0; r < 8; ++r)
      h_lds[(hi * 8 + r) * AS + colG] = f2h(hv[r]);        // publish h_t for step t+1
  }
}

// ---------------------------------------------------------------------------
extern "C" void kernel_launch(void* const* d_in, const int* in_sizes, int n_in,
                              void* d_out, int out_size, void* d_ws, size_t ws_size,
                              hipStream_t stream) {
  (void)in_sizes; (void)n_in; (void)out_size; (void)ws_size;
  const int*   x    = (const int*)d_in[0];
  const float* emb  = (const float*)d_in[1];
  const float* Whh  = (const float*)d_in[2];
  const float* bhh  = (const float*)d_in[3];
  const float* Wxh  = (const float*)d_in[4];
  const float* bxh  = (const float*)d_in[5];
  float* out = (float*)d_out;

  unsigned short* wxh_h = (unsigned short*)d_ws;          // H*H f16
  unsigned short* whh_h = wxh_h + H * H;                  // H*H f16

  cvt_f16<<<(H * H) / 256, 256, 0, stream>>>(Wxh, wxh_h, H * H);
  cvt_f16<<<(H * H) / 256, 256, 0, stream>>>(Whh, whh_h, H * H);
  xh_gemm<<<(NB * LSEQ) / 16, 256, 0, stream>>>(x, emb, wxh_h, bxh, out);
  rnn_scan<<<NB / 16, 512, 0, stream>>>(whh_h, bhh, out);
}
